// VectorQuantization_66468913873517
// MI455X (gfx1250) — matrix operand order repaired and checked by reading, not compiled, not run on hardware
//
#include <hip/hip_runtime.h>

typedef __attribute__((ext_vector_type(16))) _Float16 v16h;
typedef __attribute__((ext_vector_type(8)))  _Float16 v8h;
typedef __attribute__((ext_vector_type(8)))  float    v8f;

#define B_  8
#define T_  4096
#define D_  512
#define K_  4096
#define M_  (B_ * T_)   // 32768 rows

// ---------------------------------------------------------------------------
// Kernel 0: L2-normalize each row of x (eps clamp 1e-12) and convert to f16.
// One wave (32 lanes) per row; each lane handles 16 contiguous floats.
// ---------------------------------------------------------------------------
__global__ __launch_bounds__(256) void vq_normalize_f16(
    const float* __restrict__ x, _Float16* __restrict__ xn)
{
  const int row  = blockIdx.x * 8 + (threadIdx.x >> 5);
  const int lane = threadIdx.x & 31;
  const float4* p = (const float4*)(x + (size_t)row * D_ + lane * 16);
  float4 a0 = p[0], a1 = p[1], a2 = p[2], a3 = p[3];
  float s = a0.x*a0.x + a0.y*a0.y + a0.z*a0.z + a0.w*a0.w
          + a1.x*a1.x + a1.y*a1.y + a1.z*a1.z + a1.w*a1.w
          + a2.x*a2.x + a2.y*a2.y + a2.z*a2.z + a2.w*a2.w
          + a3.x*a3.x + a3.y*a3.y + a3.z*a3.z + a3.w*a3.w;
  #pragma unroll
  for (int off = 16; off > 0; off >>= 1) s += __shfl_xor(s, off, 32);
  const float scale = 1.0f / fmaxf(sqrtf(s), 1e-12f);

  v8h h0, h1;
  h0[0]=(_Float16)(a0.x*scale); h0[1]=(_Float16)(a0.y*scale);
  h0[2]=(_Float16)(a0.z*scale); h0[3]=(_Float16)(a0.w*scale);
  h0[4]=(_Float16)(a1.x*scale); h0[5]=(_Float16)(a1.y*scale);
  h0[6]=(_Float16)(a1.z*scale); h0[7]=(_Float16)(a1.w*scale);
  h1[0]=(_Float16)(a2.x*scale); h1[1]=(_Float16)(a2.y*scale);
  h1[2]=(_Float16)(a2.z*scale); h1[3]=(_Float16)(a2.w*scale);
  h1[4]=(_Float16)(a3.x*scale); h1[5]=(_Float16)(a3.y*scale);
  h1[6]=(_Float16)(a3.z*scale); h1[7]=(_Float16)(a3.w*scale);
  _Float16* dst = xn + (size_t)row * D_ + lane * 16;
  *(v8h*)(dst)     = h0;
  *(v8h*)(dst + 8) = h1;
}

// ---------------------------------------------------------------------------
// Kernel 1: convert codebook to f16 and compute ||e||^2 (fp32).
// ---------------------------------------------------------------------------
__global__ __launch_bounds__(256) void vq_embed_f16(
    const float* __restrict__ e, _Float16* __restrict__ ef,
    float* __restrict__ esq)
{
  const int row  = blockIdx.x * 8 + (threadIdx.x >> 5);
  const int lane = threadIdx.x & 31;
  const float4* p = (const float4*)(e + (size_t)row * D_ + lane * 16);
  float4 a0 = p[0], a1 = p[1], a2 = p[2], a3 = p[3];
  float s = a0.x*a0.x + a0.y*a0.y + a0.z*a0.z + a0.w*a0.w
          + a1.x*a1.x + a1.y*a1.y + a1.z*a1.z + a1.w*a1.w
          + a2.x*a2.x + a2.y*a2.y + a2.z*a2.z + a2.w*a2.w
          + a3.x*a3.x + a3.y*a3.y + a3.z*a3.z + a3.w*a3.w;
  #pragma unroll
  for (int off = 16; off > 0; off >>= 1) s += __shfl_xor(s, off, 32);
  if (lane == 0) esq[row] = s;

  v8h h0, h1;
  h0[0]=(_Float16)a0.x; h0[1]=(_Float16)a0.y; h0[2]=(_Float16)a0.z; h0[3]=(_Float16)a0.w;
  h0[4]=(_Float16)a1.x; h0[5]=(_Float16)a1.y; h0[6]=(_Float16)a1.z; h0[7]=(_Float16)a1.w;
  h1[0]=(_Float16)a2.x; h1[1]=(_Float16)a2.y; h1[2]=(_Float16)a2.z; h1[3]=(_Float16)a2.w;
  h1[4]=(_Float16)a3.x; h1[5]=(_Float16)a3.y; h1[6]=(_Float16)a3.z; h1[7]=(_Float16)a3.w;
  _Float16* dst = ef + (size_t)row * D_ + lane * 16;
  *(v8h*)(dst)     = h0;
  *(v8h*)(dst + 8) = h1;
}

// ---------------------------------------------------------------------------
// Kernel 2: WMMA GEMM + fused argmax.
// Block = 256 threads = 8 waves; each wave owns 16 rows, preloads its whole
// A panel (16 x 512 f16 = 16 v16h fragments = 128 VGPRs), then sweeps the
// codebook TWO 16-column tiles at a time: 32 x v_wmma_f32_16x16x32_f16 per
// iteration split across two independent accumulator chains (XDL ILP),
// fused running argmax on score = 2*dot - ||e||^2.
// ---------------------------------------------------------------------------
__global__ __launch_bounds__(256) void vq_argmax_wmma(
    const _Float16* __restrict__ xn, const _Float16* __restrict__ ef,
    const float* __restrict__ esq, int* __restrict__ ind)
{
  const int lane    = threadIdx.x & 31;
  const int wave    = threadIdx.x >> 5;
  const int rowbase = blockIdx.x * 128 + wave * 16;
  const int m  = lane & 15;   // row within tile (A) / column within tile (B,C)
  const int hi = lane >> 4;   // which K-half this lane holds

  // ---- preload A fragments (row-major xn, per-lane contiguous 16B chunks)
  v16h afr[16];
  const _Float16* arow = xn + (size_t)(rowbase + m) * D_;
  #pragma unroll
  for (int s = 0; s < 16; ++s) {
    v8h lo = *(const v8h*)(arow + s * 32 + hi * 16);
    v8h hh = *(const v8h*)(arow + s * 32 + hi * 16 + 8);
    afr[s] = __builtin_shufflevector(lo, hh, 0,1,2,3,4,5,6,7,
                                             8,9,10,11,12,13,14,15);
  }

  float bestS[8];
  int   bestI[8];
  #pragma unroll
  for (int i = 0; i < 8; ++i) { bestS[i] = -3.0e38f; bestI[i] = 0; }

  for (int nt = 0; nt < K_ / 16; nt += 2) {
    const int col0 = nt * 16 + m;                // codebook row, tile 0
    const int col1 = col0 + 16;                  // codebook row, tile 1
    const _Float16* brow0 = ef + (size_t)col0 * D_ + hi * 16;
    const _Float16* brow1 = brow0 + (size_t)16 * D_;
    v8f acc0 = {};
    v8f acc1 = {};
    #pragma unroll
    for (int s = 0; s < 16; ++s) {
      v8h lo0 = *(const v8h*)(brow0 + s * 32);
      v8h hh0 = *(const v8h*)(brow0 + s * 32 + 8);
      v8h lo1 = *(const v8h*)(brow1 + s * 32);
      v8h hh1 = *(const v8h*)(brow1 + s * 32 + 8);
      v16h b0 = __builtin_shufflevector(lo0, hh0, 0,1,2,3,4,5,6,7,
                                                  8,9,10,11,12,13,14,15);
      v16h b1 = __builtin_shufflevector(lo1, hh1, 0,1,2,3,4,5,6,7,
                                                  8,9,10,11,12,13,14,15);
      acc0 = __builtin_amdgcn_wmma_f32_16x16x32_f16(
                 false, afr[s], false, b0, (short)0, acc0, false, false);
      acc1 = __builtin_amdgcn_wmma_f32_16x16x32_f16(
                 false, afr[s], false, b1, (short)0, acc1, false, false);
    }
    const float e20 = esq[col0];
    const float e21 = esq[col1];
    #pragma unroll
    for (int i = 0; i < 8; ++i) {
      const float sc0 = 2.0f * acc0[i] - e20;
      if (sc0 > bestS[i]) { bestS[i] = sc0; bestI[i] = col0; }
      const float sc1 = 2.0f * acc1[i] - e21;
      if (sc1 > bestS[i]) { bestS[i] = sc1; bestI[i] = col1; }
    }
  }

  // ---- reduce across the 16 lane-columns inside each 16-lane half.
  // C/D layout: VGPR i -> (M=i, lanes 0-15) and (M=i+8, lanes 16-31).
  #pragma unroll
  for (int i = 0; i < 8; ++i) {
    float bs = bestS[i];
    int   bi = bestI[i];
    #pragma unroll
    for (int off = 1; off < 16; off <<= 1) {
      const float os = __shfl_xor(bs, off, 16);
      const int   oi = __shfl_xor(bi, off, 16);
      if (os > bs || (os == bs && oi < bi)) { bs = os; bi = oi; }
    }
    if (m == 0) ind[rowbase + hi * 8 + i] = bi;
  }
}

// ---------------------------------------------------------------------------
// Kernel 3: quantize = embed[ind] transposed to [B, D, T].
// t contiguous across threadIdx.x -> coalesced stores; gather reads hit L2.
// ---------------------------------------------------------------------------
__global__ __launch_bounds__(256) void vq_gather(
    const float* __restrict__ embed, const int* __restrict__ ind,
    float* __restrict__ outq)
{
  const int t = blockIdx.x * 64 + (threadIdx.x & 63);
  const int d = blockIdx.y * 4 + (threadIdx.x >> 6);
  const int b = blockIdx.z;
  const int id = ind[b * T_ + t];
  outq[((size_t)(b * D_ + d)) * T_ + t] = embed[(size_t)id * D_ + d];
}

// ---------------------------------------------------------------------------
extern "C" void kernel_launch(void* const* d_in, const int* in_sizes, int n_in,
                              void* d_out, int out_size, void* d_ws, size_t ws_size,
                              hipStream_t stream) {
  const float* x     = (const float*)d_in[0];   // [B, T, D] f32
  const float* embed = (const float*)d_in[1];   // [K, D]    f32

  int*   ind  = (int*)d_out;                    // output 0: [B, T] int32
  float* outq = (float*)d_out + M_;             // output 1: [B, D, T] f32

  char* ws = (char*)d_ws;
  _Float16* xn  = (_Float16*)ws;                                  // 32 MB
  _Float16* ef  = (_Float16*)(ws + (size_t)M_ * D_ * 2);          //  4 MB
  float*    esq = (float*)   (ws + (size_t)M_ * D_ * 2
                                 + (size_t)K_ * D_ * 2);          // 16 KB

  vq_normalize_f16<<<M_ / 8, 256, 0, stream>>>(x, xn);
  vq_embed_f16<<<K_ / 8, 256, 0, stream>>>(embed, ef, esq);
  vq_argmax_wmma<<<M_ / 128, 256, 0, stream>>>(xn, ef, esq, ind);
  dim3 g(T_ / 64, D_ / 4, B_);
  vq_gather<<<g, 256, 0, stream>>>(embed, ind, outq);
}
